// Net_41566693491302
// MI455X (gfx1250) — matrix operand order repaired and checked by reading, not compiled, not run on hardware
//
#include <hip/hip_runtime.h>
#include <math.h>

// ---------------------------------------------------------------------------
// GAT (3 layers) + global max pool + FC + log_softmax for gfx1250 (CDNA5).
// GEMMs use v_wmma_f32_16x16x32_f16 (f16 in, f32 accumulate).
// Edge softmax/aggregation use native f32 atomics.
// ---------------------------------------------------------------------------

typedef __attribute__((ext_vector_type(16))) _Float16 v16h;
typedef __attribute__((ext_vector_type(8)))  float    v8f;

#define NEG_SLOPE 0.2f

// ---------------- float atomic max via int/uint atomics (native insts) ------
__device__ __forceinline__ void atomicMaxF32(float* addr, float val) {
  if (val >= 0.0f) {
    atomicMax((int*)addr, __float_as_int(val));
  } else {
    atomicMin((unsigned int*)addr, (unsigned int)__float_as_int(val));
  }
}

// ---------------- WMMA GEMM: H[n, FOUT] = X[n, KSRC] @ W[KSRC, FOUT] --------
// KPAD = KSRC rounded up to multiple of 32 (zero-padded fragments).
// One wave per 16-row tile; wave covers all FOUT/16 column tiles.
template <int KPAD, int KSRC, int FOUT>
__global__ __launch_bounds__(128) void gemm_wmma_kernel(
    const float* __restrict__ X, const float* __restrict__ W,
    float* __restrict__ H, int nrows) {
  const int lane = threadIdx.x & 31;
  const int wave = threadIdx.x >> 5;
  const int tile = blockIdx.x * 4 + wave;        // 16-row tile index
  const int rowBase = tile * 16;
  if (rowBase >= nrows) return;                  // wave-uniform exit (EXEC all-1)

  constexpr int NCT = FOUT / 16;                 // column tiles
  v8f acc[NCT];
#pragma unroll
  for (int t = 0; t < NCT; ++t) acc[t] = (v8f){0.f,0.f,0.f,0.f,0.f,0.f,0.f,0.f};

  const bool hi = (lane >= 16);
  const int m = rowBase + (lane & 15);           // A row for this lane

#pragma unroll
  for (int k0 = 0; k0 < KPAD; k0 += 32) {
    // ---- A fragment: 16x32 f16. lane<16: K k0+0..15 ; lane>=16: K k0+16..31
    const int kA0 = k0 + (hi ? 16 : 0);
    const int kA1 = kA0 + 8;
    const float* xr = X + (size_t)m * KSRC;
    v16h a;
#pragma unroll
    for (int j = 0; j < 8; ++j) {
      const int k1 = kA0 + j, k2 = kA1 + j;
      a[j]     = (k1 < KSRC) ? (_Float16)xr[k1] : (_Float16)0.f;
      a[j + 8] = (k2 < KSRC) ? (_Float16)xr[k2] : (_Float16)0.f;
    }
#pragma unroll
    for (int t = 0; t < NCT; ++t) {
      // ---- B fragment: 32x16 f16. lanes0-15: K k0..k0+15 of col=lane ;
      //      lanes16-31: K k0+16..k0+31 of col=lane-16.
      const int col = t * 16 + (lane & 15);
      const int kB  = k0 + (hi ? 16 : 0);
      v16h b;
#pragma unroll
      for (int j = 0; j < 16; ++j) {
        const int kk = kB + j;
        b[j] = (kk < KSRC) ? (_Float16)W[(size_t)kk * FOUT + col] : (_Float16)0.f;
      }
      acc[t] = __builtin_amdgcn_wmma_f32_16x16x32_f16(
          /*neg_a=*/false, a, /*neg_b=*/false, b,
          /*c_mod=*/(short)0, acc[t], /*reuse_a=*/false, /*reuse_b=*/false);
    }
  }

  // ---- store D: VGPR v -> M = v + 8*(lane>=16), N = lane&15
#pragma unroll
  for (int t = 0; t < NCT; ++t) {
    const int col = t * 16 + (lane & 15);
#pragma unroll
    for (int v = 0; v < 8; ++v) {
      const int row = rowBase + v + (hi ? 8 : 0);
      H[(size_t)row * FOUT + col] = acc[t][v];
    }
  }
}

// ---------------- per-node attention scalars s = h.a_src, d = h.a_dst -------
template <int F>
__global__ void node_sd_kernel(const float* __restrict__ H,
                               const float* __restrict__ a_src,
                               const float* __restrict__ a_dst,
                               float* __restrict__ S, float* __restrict__ D, int n) {
  const int i = blockIdx.x * blockDim.x + threadIdx.x;
  if (i >= n) return;
  const float* h = H + (size_t)i * F;
  float s = 0.f, d = 0.f;
#pragma unroll
  for (int f = 0; f < F; ++f) { const float hv = h[f]; s += hv * a_src[f]; d += hv * a_dst[f]; }
  S[i] = s; D[i] = d;
}

// ---------------- fills -----------------------------------------------------
__global__ void fill_kernel(float* __restrict__ p, float v, int n) {
  const int i = blockIdx.x * blockDim.x + threadIdx.x;
  if (i < n) p[i] = v;
}

// ---------------- edge helpers ----------------------------------------------
__device__ __forceinline__ void edge_uv(const int* __restrict__ src,
                                        const int* __restrict__ dst,
                                        int e, int E, int& u, int& v) {
  if (e < E) { u = src[e]; v = dst[e]; } else { u = e - E; v = e - E; }
}

__device__ __forceinline__ float lrelu(float x) {
  return x > 0.f ? x : NEG_SLOPE * x;
}

// pass 1: per-dst segment max of leaky_relu(s[src]+d[dst])
__global__ void edge_max_kernel(const int* __restrict__ src, const int* __restrict__ dst,
                                const float* __restrict__ S, const float* __restrict__ D,
                                float* __restrict__ M, int E, int ET) {
  const int e = blockIdx.x * blockDim.x + threadIdx.x;
  if (e >= ET) return;
  int u, v; edge_uv(src, dst, e, E, u, v);
  atomicMaxF32(&M[v], lrelu(S[u] + D[v]));
}

// pass 2: denom[dst] += exp(l - m[dst])
__global__ void edge_expsum_kernel(const int* __restrict__ src, const int* __restrict__ dst,
                                   const float* __restrict__ S, const float* __restrict__ D,
                                   const float* __restrict__ M, float* __restrict__ DEN,
                                   int E, int ET) {
  const int e = blockIdx.x * blockDim.x + threadIdx.x;
  if (e >= ET) return;
  int u, v; edge_uv(src, dst, e, E, u, v);
  atomicAdd(&DEN[v], __expf(lrelu(S[u] + D[v]) - M[v]));
}

// pass 3: out[dst] += alpha * h[src]
template <int F>
__global__ void edge_aggregate_kernel(const int* __restrict__ src, const int* __restrict__ dst,
                                      const float* __restrict__ S, const float* __restrict__ D,
                                      const float* __restrict__ M, const float* __restrict__ DEN,
                                      const float* __restrict__ H, float* __restrict__ OUT,
                                      int E, int ET) {
  const int e = blockIdx.x * blockDim.x + threadIdx.x;
  if (e >= ET) return;
  int u, v; edge_uv(src, dst, e, E, u, v);
  const float ex = __expf(lrelu(S[u] + D[v]) - M[v]);
  const float alpha = ex / (DEN[v] + 1e-16f);
  const float* hs = H + (size_t)u * F;
  float* ov = OUT + (size_t)v * F;
#pragma unroll
  for (int f = 0; f < F; f += 4) {
    const float4 hv = *(const float4*)(hs + f);
    atomicAdd(ov + f + 0, alpha * hv.x);
    atomicAdd(ov + f + 1, alpha * hv.y);
    atomicAdd(ov + f + 2, alpha * hv.z);
    atomicAdd(ov + f + 3, alpha * hv.w);
  }
}

// pass 4 (in place): x = relu(agg + b)
template <int F>
__global__ void bias_relu_kernel(float* __restrict__ A, const float* __restrict__ B, int total) {
  const int i = blockIdx.x * blockDim.x + threadIdx.x;
  if (i >= total) return;
  A[i] = fmaxf(A[i] + B[i % F], 0.f);
}

// ---------------- global max pool by batch ----------------------------------
__global__ void pool_max_kernel(const float* __restrict__ H, const int* __restrict__ batch,
                                float* __restrict__ pooled, int n) {
  const int i = blockIdx.x * blockDim.x + threadIdx.x;
  const int node = i >> 6, f = i & 63;
  if (node >= n) return;
  atomicMaxF32(&pooled[(size_t)batch[node] * 64 + f], H[(size_t)node * 64 + f]);
}

// ---------------- FC + log_softmax ------------------------------------------
__global__ void fc_logsoftmax_kernel(const float* __restrict__ pooled,
                                     const float* __restrict__ fcw,
                                     const float* __restrict__ fcb,
                                     float* __restrict__ out, int G, int C) {
  const int g = blockIdx.x * blockDim.x + threadIdx.x;
  if (g >= G) return;
  float logits[16];
  const float* p = pooled + (size_t)g * 64;
  for (int c = 0; c < C; ++c) {
    float acc = fcb[c];
#pragma unroll
    for (int f = 0; f < 64; ++f) acc += p[f] * fcw[f * C + c];
    logits[c] = acc;
  }
  float mx = logits[0];
  for (int c = 1; c < C; ++c) mx = fmaxf(mx, logits[c]);
  float sum = 0.f;
  for (int c = 0; c < C; ++c) sum += __expf(logits[c] - mx);
  const float lse = mx + __logf(sum);
  for (int c = 0; c < C; ++c) out[(size_t)g * C + c] = logits[c] - lse;
}

// ---------------------------------------------------------------------------
static inline int ceil_div(int a, int b) { return (a + b - 1) / b; }

extern "C" void kernel_launch(void* const* d_in, const int* in_sizes, int n_in,
                              void* d_out, int out_size, void* d_ws, size_t ws_size,
                              hipStream_t stream) {
  // inputs (setup_inputs order)
  const float* x      = (const float*)d_in[0];
  const int*   ei     = (const int*)  d_in[1];
  const int*   batch  = (const int*)  d_in[2];
  const float* W1     = (const float*)d_in[3];
  const float* a1s    = (const float*)d_in[4];
  const float* a1d    = (const float*)d_in[5];
  const float* b1     = (const float*)d_in[6];
  const float* W2     = (const float*)d_in[7];
  const float* a2s    = (const float*)d_in[8];
  const float* a2d    = (const float*)d_in[9];
  const float* b2     = (const float*)d_in[10];
  const float* W3     = (const float*)d_in[11];
  const float* a3s    = (const float*)d_in[12];
  const float* a3d    = (const float*)d_in[13];
  const float* b3     = (const float*)d_in[14];
  const float* fcw    = (const float*)d_in[15];
  const float* fcb    = (const float*)d_in[16];
  float*       out    = (float*)d_out;

  const int N  = in_sizes[2];          // batch vector length == #nodes
  const int E  = in_sizes[1] / 2;      // edge_index is (2, E)
  const int C  = in_sizes[16];         // fc_b length
  const int G  = out_size / C;
  const int ET = E + N;                // edges + self loops

  const int* src = ei;
  const int* dst = ei + E;

  // workspace layout (floats), 256B-aligned chunks
  char* ws = (char*)d_ws;
  size_t off = 0;
  auto alloc = [&](size_t nfloats) {
    float* p = (float*)(ws + off);
    off += ((nfloats * sizeof(float)) + 255) & ~(size_t)255;
    return p;
  };
  float* P    = alloc((size_t)N * 64);   // feature ping
  float* Q    = alloc((size_t)N * 64);   // feature pong
  float* HL   = alloc((size_t)N * 64);   // h = x @ W
  float* S    = alloc(N);
  float* D    = alloc(N);
  float* M    = alloc(N);
  float* DEN  = alloc(N);
  float* POOL = alloc((size_t)G * 64);
  (void)ws_size;

  const int TB = 256;
  const int gemmWavesPerBlock = 4;

  auto run_layer = [&](auto kpad_t, auto ksrc_t, auto fout_t,
                       const float* xin, const float* W,
                       const float* as, const float* ad, const float* bb,
                       float* agg) {
    constexpr int KPAD = decltype(kpad_t)::value;
    constexpr int KSRC = decltype(ksrc_t)::value;
    constexpr int FOUT = decltype(fout_t)::value;

    const int tiles = ceil_div(N, 16);
    gemm_wmma_kernel<KPAD, KSRC, FOUT>
        <<<ceil_div(tiles, gemmWavesPerBlock), 32 * gemmWavesPerBlock, 0, stream>>>(xin, W, HL, N);
    node_sd_kernel<FOUT><<<ceil_div(N, TB), TB, 0, stream>>>(HL, as, ad, S, D, N);
    fill_kernel<<<ceil_div(N, TB), TB, 0, stream>>>(M, -INFINITY, N);
    fill_kernel<<<ceil_div(N, TB), TB, 0, stream>>>(DEN, 0.f, N);
    fill_kernel<<<ceil_div(N * FOUT, TB), TB, 0, stream>>>(agg, 0.f, N * FOUT);
    edge_max_kernel<<<ceil_div(ET, TB), TB, 0, stream>>>(src, dst, S, D, M, E, ET);
    edge_expsum_kernel<<<ceil_div(ET, TB), TB, 0, stream>>>(src, dst, S, D, M, DEN, E, ET);
    edge_aggregate_kernel<FOUT><<<ceil_div(ET, TB), TB, 0, stream>>>(src, dst, S, D, M, DEN, HL, agg, E, ET);
    bias_relu_kernel<FOUT><<<ceil_div(N * FOUT, TB), TB, 0, stream>>>(agg, bb, N * FOUT);
  };

  using i128 = std::integral_constant<int, 128>;
  using i64  = std::integral_constant<int, 64>;
  using i32  = std::integral_constant<int, 32>;
  using i16  = std::integral_constant<int, 16>;

  // layer 1: 128 -> 16   (K=128)
  run_layer(i128{}, i128{}, i16{}, x, W1, a1s, a1d, b1, P);
  // layer 2: 16 -> 32    (K=16 padded to 32)
  run_layer(i32{}, i16{}, i32{}, P, W2, a2s, a2d, b2, Q);
  // layer 3: 32 -> 64    (K=32)
  run_layer(i32{}, i32{}, i64{}, Q, W3, a3s, a3d, b3, P);

  // global max pool over batch -> [G, 64]
  fill_kernel<<<ceil_div(G * 64, TB), TB, 0, stream>>>(POOL, -INFINITY, G * 64);
  pool_max_kernel<<<ceil_div(N * 64, TB), TB, 0, stream>>>(P, batch, POOL, N);

  // FC + log_softmax -> [G, C]
  fc_logsoftmax_kernel<<<ceil_div(G, TB), TB, 0, stream>>>(POOL, fcw, fcb, out, G, C);
}
